// AttentionActor_9517647528446
// MI455X (gfx1250) — compile-verified, matrix operand order checked by reading, not run on hardware
//
#include <hip/hip_runtime.h>

// ---------------------------------------------------------------------------
// Types for CDNA5 WMMA (wave32)
// ---------------------------------------------------------------------------
typedef __bf16 bf16;
typedef __attribute__((ext_vector_type(16))) __bf16 v16bf;
typedef __attribute__((ext_vector_type(8)))  __bf16 v8bf;
typedef __attribute__((ext_vector_type(8)))  float  v8f;
typedef __attribute__((ext_vector_type(4)))  unsigned int u32x4;
typedef __attribute__((ext_vector_type(8)))  int          i32x8;
typedef __attribute__((ext_vector_type(4)))  int          i32x4;

#define TAU_INV 0.5f
#define THRESH  0.5f
#define T_STEPS 16

// Problem sizes (fixed by the reference)
#define BATCH 8192
#define INDIM 1024
#define HID   2048
#define ODIM  512

// Use the Tensor Data Mover when the toolchain exposes it (device pass only).
#if defined(__gfx1250__) && __has_builtin(__builtin_amdgcn_tensor_load_to_lds)
#define USE_TDM 1
#else
#define USE_TDM 0
#endif

// ---------------------------------------------------------------------------
// Elementwise helpers
// ---------------------------------------------------------------------------
__global__ void zero_f32_kernel(float* __restrict__ p, size_t n) {
    size_t i = (size_t)blockIdx.x * blockDim.x + threadIdx.x;
    size_t stride = (size_t)gridDim.x * blockDim.x;
    for (; i < n; i += stride) p[i] = 0.0f;
}

__global__ void f32_to_bf16_kernel(const float* __restrict__ src,
                                   bf16* __restrict__ dst, size_t n) {
    size_t i = (size_t)blockIdx.x * blockDim.x + threadIdx.x;
    size_t stride = (size_t)gridDim.x * blockDim.x;
    for (; i < n; i += stride) dst[i] = (bf16)src[i];
}

// src: f32 [K][N] row-major  ->  dst: bf16 [N][K] row-major (K contiguous)
__global__ __launch_bounds__(256)
void transpose_to_bf16_kernel(const float* __restrict__ src,
                              bf16* __restrict__ dst, int K, int N) {
    __shared__ float tile[32][33];
    int nBase = blockIdx.x * 32;
    int kBase = blockIdx.y * 32;
    for (int j = threadIdx.y; j < 32; j += 8)
        tile[j][threadIdx.x] = src[(size_t)(kBase + j) * N + nBase + threadIdx.x];
    __syncthreads();
    for (int j = threadIdx.y; j < 32; j += 8)
        dst[(size_t)(nBase + j) * K + kBase + threadIdx.x] =
            (bf16)tile[threadIdx.x][j];
}

// Input-node LIF: mem1 += (attn - mem1)/tau; spike; hard reset. Writes s1 (bf16).
__global__ void lif_input_kernel(const float* __restrict__ attn,
                                 float* __restrict__ mem1,
                                 bf16* __restrict__ s1, size_t n) {
    size_t i = (size_t)blockIdx.x * blockDim.x + threadIdx.x;
    size_t stride = (size_t)gridDim.x * blockDim.x;
    for (; i < n; i += stride) {
        float m = mem1[i];
        m += (attn[i] - m) * TAU_INV;
        float s = (m > THRESH) ? 1.0f : 0.0f;
        mem1[i] = m * (1.0f - s);
        s1[i]   = (bf16)s;
    }
}

// ---------------------------------------------------------------------------
// Tensor Data Mover: DMA one [rows x 64] bf16 tile (row pitch 144B in LDS:
// 128B data + 16B pad -> matches PITCH=72 elements, bank-conflict friendly).
// D# layout per CDNA5 ISA ch.8 (async/tensor).
// ---------------------------------------------------------------------------
#if USE_TDM
__device__ __forceinline__ void tdm_load_tile(const bf16* gsrc, unsigned ldsOff,
                                              int K, int rows) {
    unsigned long long ga = (unsigned long long)(size_t)gsrc;
    u32x4 g0;
    g0[0] = 1u;                                   // count=1 (valid user D#)
    g0[1] = ldsOff;                               // lds_addr (bytes)
    g0[2] = (unsigned)(ga & 0xFFFFFFFFu);         // global_addr[31:0]
    g0[3] = (unsigned)((ga >> 32) & 0x1FFFFFFu)   // global_addr[56:32]
          | (2u << 30);                           // type = 2 ("image")
    i32x8 g1;
    g1[0] = (int)((1u << 16)      // data_size = 2 bytes
                | (1u << 20)      // pad_enable
                | (4u << 22)      // pad_interval: 16*8B = 128B of data
                | (3u << 25));    // pad_amount : 4 DWORDs = 16B
    g1[1] = (int)(((unsigned)K & 0xFFFFu) << 16);        // tensor_dim0[15:0]
    g1[2] = (int)((((unsigned)K >> 16) & 0xFFFFu)        // tensor_dim0[31:16]
                | (0xFFFFu << 16));                      // tensor_dim1[15:0] (big)
    g1[3] = (int)(64u << 16);                            // tile_dim0 = 64
    g1[4] = (int)((unsigned)rows & 0xFFFFu);             // tile_dim1; tile_dim2=0
    g1[5] = K;                                           // tensor_dim0_stride[31:0]
    g1[6] = 0;
    g1[7] = 0;
    i32x4 z4 = {0, 0, 0, 0};
#if __clang_major__ >= 23
    i32x8 z8 = {0, 0, 0, 0, 0, 0, 0, 0};
    __builtin_amdgcn_tensor_load_to_lds(g0, g1, z4, z4, z8, 0);
#else
    __builtin_amdgcn_tensor_load_to_lds(g0, g1, z4, z4, 0);
#endif
}
#endif

// ---------------------------------------------------------------------------
// Fused GEMM:  C[M,N] = A[M,K](bf16) @ Bt[N,K](bf16)^T + bias, with epilogue.
//   EPI 0: outF = C
//   EPI 1: LIF on memBuf -> spikeOut (bf16), hard reset
//   EPI 2: leaky integrate memBuf += (C - memBuf)/tau
// Block tile 128x128, K-step 64; 8 waves (wave32) in a 4(M) x 2(N) grid,
// each wave computes 32x64 via 16x v_wmma_f32_16x16x32_bf16 per K-step.
// TDM path: double-buffered LDS, wave0 issues tensor_load_to_lds for stage
// k+1 right after the barrier so DMA overlaps WMMA; one barrier per K-step.
// ---------------------------------------------------------------------------
template <int EPI>
__global__ __launch_bounds__(256)
void gemm_bf16_wmma(const bf16* __restrict__ A, const bf16* __restrict__ Bt,
                    const float* __restrict__ bias,
                    float* __restrict__ memBuf, bf16* __restrict__ spikeOut,
                    float* __restrict__ outF, int N, int K) {
    constexpr int BM = 128, BN = 128, BK = 64;
    constexpr int PITCH = 72;  // 144B row pitch (128B data + 16B pad)
#if USE_TDM
    constexpr int STAGES = 2;
#else
    constexpr int STAGES = 1;
#endif
    __shared__ __align__(16) bf16 As[STAGES][BM][PITCH];
    __shared__ __align__(16) bf16 Bs[STAGES][BN][PITCH];

    const int t    = threadIdx.x;
    const int lane = t & 31;
    const int wave = t >> 5;
    const int wm   = wave & 3;   // wave row -> 32 rows of C
    const int wn   = wave >> 2;  // wave col -> 64 cols of C
    const int half = lane >> 4;  // K-half select per ISA A/B fragment layout
    const int l16  = lane & 15;

    const size_t blockM = (size_t)blockIdx.y * BM;
    const size_t blockN = (size_t)blockIdx.x * BN;
    const size_t Ks = (size_t)K;

    v8f acc[2][4] = {};
    const int kIters = K / BK;

#if USE_TDM
    const bool issuer = (wave == 0);
    if (issuer) {
        tdm_load_tile(&A [blockM * Ks], (unsigned)(size_t)&As[0][0][0], K, BM);
        tdm_load_tile(&Bt[blockN * Ks], (unsigned)(size_t)&Bs[0][0][0], K, BN);
    }
#endif

    for (int kt = 0; kt < kIters; ++kt) {
#if USE_TDM
        const int cur = kt & 1;
        if (issuer) __builtin_amdgcn_s_wait_tensorcnt(0);  // stage `cur` landed
        __syncthreads();  // publish stage `cur`; stage `cur^1` fully consumed
        if (issuer && (kt + 1 < kIters)) {
            tdm_load_tile(&A [blockM * Ks + (size_t)(kt + 1) * BK],
                          (unsigned)(size_t)&As[cur ^ 1][0][0], K, BM);
            tdm_load_tile(&Bt[blockN * Ks + (size_t)(kt + 1) * BK],
                          (unsigned)(size_t)&Bs[cur ^ 1][0][0], K, BN);
        }
#else
        const int cur = 0;
        const int k0 = kt * BK;
        __syncthreads();
        // Stage 128x64 bf16 tiles of A and Bt into LDS (16B chunks).
        for (int i = t; i < (BM * BK) / 8; i += 256) {
            int row = i >> 3;
            int seg = i & 7;
            const bf16* ga = &A [(blockM + row) * Ks + k0 + seg * 8];
            const bf16* gb = &Bt[(blockN + row) * Ks + k0 + seg * 8];
            *(uint4*)&As[0][row][seg * 8] = *(const uint4*)ga;
            *(uint4*)&Bs[0][row][seg * 8] = *(const uint4*)gb;
            if (kt + 1 < kIters) {
                __builtin_prefetch(ga + BK, 0, 1);  // global_prefetch_b8
                __builtin_prefetch(gb + BK, 0, 1);
            }
        }
        __syncthreads();
#endif

        // Fragments per the CDNA5 16-bit A/B VGPR layout (per 32-wide K slab):
        // lane<16 : K {0..7} then {16..23}; lane>=16 : K {8..15} then {24..31}
#pragma unroll
        for (int kk = 0; kk < 2; ++kk) {
            const int klo = kk * 32 + half * 8;
            v16bf af[2], bfr[4];
#pragma unroll
            for (int mt = 0; mt < 2; ++mt) {
                int r = wm * 32 + mt * 16 + l16;
                v8bf lo = *(const v8bf*)&As[cur][r][klo];
                v8bf hi = *(const v8bf*)&As[cur][r][klo + 16];
                af[mt] = __builtin_shufflevector(lo, hi, 0, 1, 2, 3, 4, 5, 6, 7,
                                                 8, 9, 10, 11, 12, 13, 14, 15);
            }
#pragma unroll
            for (int nt = 0; nt < 4; ++nt) {
                int c = wn * 64 + nt * 16 + l16;
                v8bf lo = *(const v8bf*)&Bs[cur][c][klo];
                v8bf hi = *(const v8bf*)&Bs[cur][c][klo + 16];
                bfr[nt] = __builtin_shufflevector(lo, hi, 0, 1, 2, 3, 4, 5, 6, 7,
                                                  8, 9, 10, 11, 12, 13, 14, 15);
            }
#pragma unroll
            for (int mt = 0; mt < 2; ++mt)
#pragma unroll
                for (int nt = 0; nt < 4; ++nt)
                    acc[mt][nt] = __builtin_amdgcn_wmma_f32_16x16x32_bf16(
                        false, af[mt], false, bfr[nt], (short)0, acc[mt][nt],
                        false, false);
        }
    }

    // Epilogue. C layout: VGPR j -> (M = j + 8*half, N = lane&15).
    const size_t Nl = (size_t)N;
#pragma unroll
    for (int mt = 0; mt < 2; ++mt) {
#pragma unroll
        for (int nt = 0; nt < 4; ++nt) {
            size_t gcol = blockN + wn * 64 + nt * 16 + l16;
            float b = bias[gcol];
#pragma unroll
            for (int j = 0; j < 8; ++j) {
                size_t grow = blockM + wm * 32 + mt * 16 + half * 8 + j;
                size_t idx  = grow * Nl + gcol;
                float v = acc[mt][nt][j] + b;
                if (EPI == 0) {
                    outF[idx] = v;
                } else if (EPI == 1) {
                    float m = memBuf[idx];
                    m += (v - m) * TAU_INV;
                    float s = (m > THRESH) ? 1.0f : 0.0f;
                    memBuf[idx]   = m * (1.0f - s);
                    spikeOut[idx] = (bf16)s;
                } else {
                    float m = memBuf[idx];
                    memBuf[idx] = m + (v - m) * TAU_INV;
                }
            }
        }
    }
}

// ---------------------------------------------------------------------------
// Gumbel-max sampler + one-hot. Deterministic counter-based noise.
// ---------------------------------------------------------------------------
__device__ __forceinline__ float gumbel_noise(unsigned long long i) {
    unsigned long long z = (i + 42ull) * 0x9E3779B97F4A7C15ull;
    z ^= z >> 30; z *= 0xBF58476D1CE4E5B9ull;
    z ^= z >> 27; z *= 0x94D049BB133111EBull;
    z ^= z >> 31;
    float u = (float)(z >> 40) * (1.0f / 16777216.0f);
    u = fmaxf(u, 1e-20f);
    return -__logf(-__logf(u));
}

__global__ __launch_bounds__(256)
void sampler_kernel(const float* __restrict__ memo,
                    float* __restrict__ action, float* __restrict__ pOut) {
    __shared__ float sv[256];
    __shared__ int   si[256];
    const int t = threadIdx.x;
    const size_t row  = blockIdx.x;
    const size_t base = row * ODIM;

    float best = -3.0e38f;
    int   bi   = 0;
    for (int c = t; c < ODIM; c += 256) {
        float v = memo[base + c] + gumbel_noise(base + c);
        if (v > best) { best = v; bi = c; }
    }
    sv[t] = best; si[t] = bi;
    __syncthreads();
    for (int off = 128; off > 0; off >>= 1) {
        if (t < off && sv[t + off] > sv[t]) { sv[t] = sv[t + off]; si[t] = si[t + off]; }
        __syncthreads();
    }
    const int win = si[0];
    for (int c = t; c < ODIM; c += 256) {
        action[base + c] = (c == win) ? 1.0f : 0.0f;
        pOut[base + c]   = memo[base + c];
    }
}

// ---------------------------------------------------------------------------
// Host-side orchestration (graph-capture safe: kernels only, all on `stream`)
// ---------------------------------------------------------------------------
extern "C" void kernel_launch(void* const* d_in, const int* in_sizes, int n_in,
                              void* d_out, int out_size, void* d_ws, size_t ws_size,
                              hipStream_t stream) {
    (void)in_sizes; (void)n_in; (void)out_size; (void)ws_size;

    const float* state = (const float*)d_in[0];  // [8192,1024]
    const float* We    = (const float*)d_in[1];  // [1024,2048]
    const float* be    = (const float*)d_in[2];  // [2048]
    const float* W1    = (const float*)d_in[3];  // [2048,2048]
    const float* b1    = (const float*)d_in[4];  // [2048]
    const float* W2    = (const float*)d_in[5];  // [2048,512]
    const float* b2    = (const float*)d_in[6];  // [512]

    // Workspace carve-up (all sizes are multiples of 16B)
    char* w = (char*)d_ws;
    bf16* stateBf = (bf16*)w;  w += (size_t)BATCH * INDIM * sizeof(bf16);
    bf16* WeT     = (bf16*)w;  w += (size_t)HID * INDIM * sizeof(bf16);
    bf16* W1T     = (bf16*)w;  w += (size_t)HID * HID * sizeof(bf16);
    bf16* W2T     = (bf16*)w;  w += (size_t)ODIM * HID * sizeof(bf16);
    bf16* s1      = (bf16*)w;  w += (size_t)BATCH * HID * sizeof(bf16);
    bf16* sf      = (bf16*)w;  w += (size_t)BATCH * HID * sizeof(bf16);
    float* attn   = (float*)w; w += (size_t)BATCH * HID * sizeof(float);
    float* mem1   = (float*)w; w += (size_t)BATCH * HID * sizeof(float);
    float* memf   = (float*)w; w += (size_t)BATCH * HID * sizeof(float);
    float* memo   = (float*)w; w += (size_t)BATCH * ODIM * sizeof(float);

    float* outAction = (float*)d_out;
    float* outP      = outAction + (size_t)BATCH * ODIM;

    // 1) Prep: convert/transpose to bf16
    {
        size_t n = (size_t)BATCH * INDIM;
        f32_to_bf16_kernel<<<8192, 256, 0, stream>>>(state, stateBf, n);
        dim3 tb(32, 8);
        transpose_to_bf16_kernel<<<dim3(HID / 32, INDIM / 32), tb, 0, stream>>>(We, WeT, INDIM, HID);
        transpose_to_bf16_kernel<<<dim3(HID / 32, HID / 32),  tb, 0, stream>>>(W1, W1T, HID, HID);
        transpose_to_bf16_kernel<<<dim3(ODIM / 32, HID / 32), tb, 0, stream>>>(W2, W2T, HID, ODIM);
    }
    // 2) Zero membrane state (mem1, memf, memo are contiguous)
    {
        size_t n = (size_t)BATCH * HID * 2 + (size_t)BATCH * ODIM;
        zero_f32_kernel<<<8192, 256, 0, stream>>>(mem1, n);
    }
    // 3) attn = state @ We + be  (plain epilogue)
    gemm_bf16_wmma<0><<<dim3(HID / 128, BATCH / 128), 256, 0, stream>>>(
        stateBf, WeT, be, nullptr, nullptr, attn, HID, INDIM);

    // 4) T timesteps
    for (int step = 0; step < T_STEPS; ++step) {
        lif_input_kernel<<<8192, 256, 0, stream>>>(attn, mem1, s1,
                                                   (size_t)BATCH * HID);
        // h = s1 @ W1 + b1 ; fused hidden LIF -> sf, memf
        gemm_bf16_wmma<1><<<dim3(HID / 128, BATCH / 128), 256, 0, stream>>>(
            s1, W1T, b1, memf, sf, nullptr, HID, HID);
        // o = sf @ W2 + b2 ; fused leaky integration -> memo
        gemm_bf16_wmma<2><<<dim3(ODIM / 128, BATCH / 128), 256, 0, stream>>>(
            sf, W2T, b2, memo, nullptr, nullptr, ODIM, HID);
    }

    // 5) Gumbel-max sample + one-hot, emit (action, p)
    sampler_kernel<<<BATCH, 256, 0, stream>>>(memo, outAction, outP);
}
